// DisplacementTensors_27127013442038
// MI455X (gfx1250) — matrix operand order repaired and checked by reading, not compiled, not run on hardware
//
#include <hip/hip_runtime.h>
#include <math.h>

typedef __attribute__((ext_vector_type(16))) _Float16 v16h;
typedef __attribute__((ext_vector_type(8)))  float    v8f;

#define DIM 32
#define WPB 8   // waves per block (256 threads, wave32)

// ---------------------------------------------------------------------------
// wave-local LDS ordering: each wave only touches its own LDS region, and
// wave32 execution is lockstep, so a DS-counter wait + scheduling barrier is
// sufficient (no s_barrier needed).
__device__ __forceinline__ void lds_sync_wave() {
  __builtin_amdgcn_wave_barrier();
  asm volatile("s_wait_dscnt 0" ::: "memory");
  __builtin_amdgcn_wave_barrier();
}

__device__ __forceinline__ float leaky(float x) { return x >= 0.f ? x : 0.1f * x; }

// Guaranteed-native fire-and-forget fp32 atomic add at L2 (no return ->
// STOREcnt-tracked; s_endpgm's implicit wait-idle covers completion).
__device__ __forceinline__ void atomic_fadd(float* p, float v) {
  asm volatile("global_atomic_add_f32 %0, %1, off" : : "v"(p), "v"(v) : "memory");
}

// B-matrix (32x16, K x N) layout for v_wmma_f32_16x16x32_f16:
// lane n (<16) holds K=0..7,16..23 of column n; lane n+16 holds K=8..15,24..31.
// Column n of B = row (cb*16+n) of W (since h = x @ W^T).
__device__ __forceinline__ v16h load_b_from_w(const float* W, int lane, int cb) {
  const int row = cb * 16 + (lane & 15);
  const int kb  = (lane < 16) ? 0 : 8;
  const float* p = W + row * DIM + kb;
  v16h r;
#pragma unroll
  for (int i = 0; i < 8; ++i) { r[i] = (_Float16)p[i]; r[8 + i] = (_Float16)p[16 + i]; }
  return r;
}

// A-matrix (16x32, M x K) layout: lane m (<16) holds K=0..7,16..23 of row m;
// lane m+16 holds K=8..15,24..31.  X is a row-major 32x32 f16 tile in LDS.
__device__ __forceinline__ v16h load_a_from_lds(const _Float16* X, int lane, int h) {
  const int m  = (lane & 15) + h * 16;
  const int kb = (lane < 16) ? 0 : 8;
  const _Float16* row = X + m * DIM;
  v16h r;
#pragma unroll
  for (int i = 0; i < 8; ++i) {
    r[i]     = row[kb + i];
    r[8 + i] = row[16 + kb + i];
  }
  return r;
}

// One MLP layer for this wave's 32x32 tile: reads cur, writes nxt (f16).
__device__ __forceinline__ void mlp_layer(const _Float16* cur, _Float16* nxt,
                                          int lane, v16h w0, v16h w1,
                                          float bv0, float bv1, bool act) {
  const int n    = lane & 15;
  const int moff = (lane < 16) ? 0 : 8;
#pragma unroll
  for (int h = 0; h < 2; ++h) {
    v16h a = load_a_from_lds(cur, lane, h);
#pragma unroll
    for (int cb = 0; cb < 2; ++cb) {
      const float bv = cb ? bv1 : bv0;
      v8f c;
#pragma unroll
      for (int j = 0; j < 8; ++j) c[j] = bv;       // bias: C[m][n] = b[n_global]
      c = __builtin_amdgcn_wmma_f32_16x16x32_f16(
          false, a, false, (cb ? w1 : w0), (short)0, c, false, false);
#pragma unroll
      for (int j = 0; j < 8; ++j) {                 // D: VGPR j -> M = j + moff
        float v = act ? leaky(c[j]) : c[j];
        nxt[(h * 16 + moff + j) * DIM + cb * 16 + n] = (_Float16)v;
      }
    }
  }
}

// ---------------------------------------------------------------------------
__global__ __launch_bounds__(256)
void zero_kernel(float* p, long n) {
  long i = (long)blockIdx.x * blockDim.x + threadIdx.x;
  long stride = (long)gridDim.x * blockDim.x;
  for (; i < n; i += stride) p[i] = 0.f;
}

// ---------------------------------------------------------------------------
// Edge kernel: one wave handles 32 edges.
__global__ __launch_bounds__(256)
void edge_kernel(const float* __restrict__ r_ij, const float* __restrict__ res_emb,
                 const int* __restrict__ src, const int* __restrict__ dst,
                 const float* __restrict__ W1, const float* __restrict__ b1,
                 const float* __restrict__ W2, const float* __restrict__ b2,
                 const float* __restrict__ W3,
                 float* __restrict__ Aa, float* __restrict__ Av, int nEdges) {
  const int wid  = threadIdx.x >> 5;
  const int lane = threadIdx.x & 31;

  __shared__ _Float16 Xs[WPB][2][32 * DIM];   // double-buffered activation tiles
  __shared__ float    rvs[WPB][32][3];        // saturated direction vectors

  const long base = ((long)blockIdx.x * WPB + wid) * 32;
  if (base + 32 > (long)nEdges) return;       // uniform per wave (E % 32 == 0)

  // Preload weights into B-layout registers (L2/const-cached, redundant per wave).
  v16h w1a = load_b_from_w(W1, lane, 0), w1b = load_b_from_w(W1, lane, 1);
  v16h w2a = load_b_from_w(W2, lane, 0), w2b = load_b_from_w(W2, lane, 1);
  v16h w3a = load_b_from_w(W3, lane, 0), w3b = load_b_from_w(W3, lane, 1);
  const float b1v0 = b1[lane & 15], b1v1 = b1[16 + (lane & 15)];
  const float b2v0 = b2[lane & 15], b2v1 = b2[16 + (lane & 15)];

  // ---- Phase 1: per-lane edge features -> LDS f16 tile -------------------
  const long e = base + lane;
  const float rx = r_ij[e * 3 + 0], ry = r_ij[e * 3 + 1], rz = r_ij[e * 3 + 2];
  const float dist = sqrtf(rx * rx + ry * ry + rz * rz);
  const float* emb = res_emb + (long)dst[e] * DIM;

  _Float16* X0 = &Xs[wid][0][0];
  const float PIR = 0.62831853071795864769f;  // pi / R0
#pragma unroll
  for (int j = 0; j < 16; ++j) {
    const float ph = PIR * (float)(1 + (j >> 1)) * dist;
    X0[lane * DIM + j]      = (_Float16)(__cosf(ph) + emb[j]);
    X0[lane * DIM + 16 + j] = (_Float16)(__sinf(ph) + emb[16 + j]);
  }
  const float nv = 1.4f * dist;               // |r * 7/R0|
  const float sc = (nv > 0.f) ? (tanhf(nv) / nv) : 0.f;
  rvs[wid][lane][0] = rx * 1.4f * sc;
  rvs[wid][lane][1] = ry * 1.4f * sc;
  rvs[wid][lane][2] = rz * 1.4f * sc;
  lds_sync_wave();

  // ---- Phase 2: 3-layer MLP via WMMA -------------------------------------
  _Float16* cur = &Xs[wid][0][0];
  _Float16* nxt = &Xs[wid][1][0];
  mlp_layer(cur, nxt, lane, w1a, w1b, b1v0, b1v1, true);
  lds_sync_wave();
  mlp_layer(nxt, cur, lane, w2a, w2b, b2v0, b2v1, true);
  lds_sync_wave();

  // ---- Phase 3: layer 3 + fused segment-sum scatter ----------------------
  const int n    = lane & 15;
  const int moff = (lane < 16) ? 0 : 8;
#pragma unroll
  for (int h = 0; h < 2; ++h) {
    v16h a = load_a_from_lds(cur, lane, h);
#pragma unroll
    for (int cb = 0; cb < 2; ++cb) {
      v8f c;
#pragma unroll
      for (int j = 0; j < 8; ++j) c[j] = 0.f;
      c = __builtin_amdgcn_wmma_f32_16x16x32_f16(
          false, a, false, (cb ? w3b : w3a), (short)0, c, false, false);
#pragma unroll
      for (int j = 0; j < 8; ++j) {
        const int  m   = h * 16 + moff + j;       // local edge index
        const long ee  = base + m;
        const int  s   = src[ee];
        const int  col = cb * 16 + n;
        const float val = c[j];
        atomic_fadd(&Aa[(long)s * DIM + col], val);
        const float r0 = rvs[wid][m][0], r1 = rvs[wid][m][1], r2 = rvs[wid][m][2];
        const long av = ((long)s * DIM + col) * 3;
        atomic_fadd(&Av[av + 0], val * r0);
        atomic_fadd(&Av[av + 1], val * r1);
        atomic_fadd(&Av[av + 2], val * r2);
      }
    }
  }
}

// ---------------------------------------------------------------------------
// Node kernel: out_v[n,v,d] = sum_a Av[n,a,d] * Wv[v,a].  One wave = 16 nodes.
__global__ __launch_bounds__(256)
void node_kernel(const float* __restrict__ Av, const float* __restrict__ Wv,
                 float* __restrict__ outv, int nNodes) {
  const int wid  = threadIdx.x >> 5;
  const int lane = threadIdx.x & 31;
  const long base = ((long)blockIdx.x * WPB + wid) * 16;
  if (base + 16 > (long)nNodes) return;       // uniform per wave (N % 16 == 0)

  v16h bv0 = load_b_from_w(Wv, lane, 0), bv1 = load_b_from_w(Wv, lane, 1);

  const int ml   = lane & 15;
  const int kb   = (lane < 16) ? 0 : 8;
  const int moff = (lane < 16) ? 0 : 8;
  const long nodeA = base + ml;

#pragma unroll
  for (int d = 0; d < 3; ++d) {
    v16h a;
#pragma unroll
    for (int i = 0; i < 8; ++i) {
      a[i]     = (_Float16)Av[(nodeA * DIM + kb + i) * 3 + d];
      a[8 + i] = (_Float16)Av[(nodeA * DIM + 16 + kb + i) * 3 + d];
    }
#pragma unroll
    for (int cb = 0; cb < 2; ++cb) {
      v8f c;
#pragma unroll
      for (int j = 0; j < 8; ++j) c[j] = 0.f;
      c = __builtin_amdgcn_wmma_f32_16x16x32_f16(
          false, a, false, (cb ? bv1 : bv0), (short)0, c, false, false);
#pragma unroll
      for (int j = 0; j < 8; ++j) {
        const long node_o = base + moff + j;
        outv[(node_o * DIM + cb * 16 + ml) * 3 + d] = c[j];
      }
    }
  }
}

// ---------------------------------------------------------------------------
extern "C" void kernel_launch(void* const* d_in, const int* in_sizes, int n_in,
                              void* d_out, int out_size, void* d_ws, size_t ws_size,
                              hipStream_t stream) {
  const float* r_ij    = (const float*)d_in[0];
  const float* res_emb = (const float*)d_in[1];
  const int*   src     = (const int*)d_in[2];
  const int*   dst     = (const int*)d_in[3];
  const float* W1      = (const float*)d_in[4];
  const float* b1      = (const float*)d_in[5];
  const float* W2      = (const float*)d_in[6];
  const float* b2      = (const float*)d_in[7];
  const float* W3      = (const float*)d_in[8];
  const float* Wv      = (const float*)d_in[9];

  const int nEdges = in_sizes[0] / 3;
  const int nNodes = in_sizes[1] / DIM;

  float* outAa = (float*)d_out;                         // [N, 32]
  float* outVv = (float*)d_out + (long)nNodes * DIM;    // [N, 32, 3]
  float* Av_ws = (float*)d_ws;                          // [N, 32, 3] accumulator

  // Zero accumulators (atomics target them).
  {
    const long nAa = (long)nNodes * DIM;
    const long nAv = (long)nNodes * DIM * 3;
    zero_kernel<<<(int)((nAa + 255) / 256), 256, 0, stream>>>(outAa, nAa);
    zero_kernel<<<(int)((nAv + 255) / 256), 256, 0, stream>>>(Av_ws, nAv);
  }

  // Edge MLP + scatter: one wave per 32 edges, 8 waves per block.
  {
    const long tiles  = ((long)nEdges + 31) / 32;
    const int  blocks = (int)((tiles + WPB - 1) / WPB);
    edge_kernel<<<blocks, 256, 0, stream>>>(r_ij, res_emb, src, dst,
                                            W1, b1, W2, b2, W3,
                                            outAa, Av_ws, nEdges);
  }

  // Node einsum: one wave per 16 nodes.
  {
    const long tiles  = ((long)nNodes + 15) / 16;
    const int  blocks = (int)((tiles + WPB - 1) / WPB);
    node_kernel<<<blocks, 256, 0, stream>>>(Av_ws, Wv, outVv, nNodes);
  }
}
